// DMFC_block_16793322127722
// MI455X (gfx1250) — matrix operand order repaired and checked by reading, not compiled, not run on hardware
//
#include <hip/hip_runtime.h>
#include <string.h>

// ---------------------------------------------------------------------------
// Problem constants (from the reference): B=16, C=128, N=2000, M=256, HEAD=4
// ---------------------------------------------------------------------------
#define B_  16
#define C_  128
#define C2_ 256   // 2*C
#define N_  2000
#define M_  256
#define H_  4
#define HD_ 32    // C/HEAD

typedef long long ll;
typedef __attribute__((ext_vector_type(16))) _Float16 v16h;
typedef __attribute__((ext_vector_type(8)))  _Float16 v8h;
typedef __attribute__((ext_vector_type(8)))  float    v8f;

// ---------------------------------------------------------------------------
// Generic strided batched GEMM on WMMA (wave32, one 16x16 tile per wave,
// 4 waves per block covering 4 adjacent n-tiles).
//   C[z, m, n] = alpha * sum_k A[z, m, k] * B[z, k, n]  (+bias[m]) (+res)
// Fragment index maps follow CDNA5 ISA 7.12.2 (f16 A 16x32 / B 32x16 /
// f32 C 16x16).  Template flags select vectorized 16B loads when the k
// dimension is unit-stride for that operand (all such call sites are
// 16-byte aligned: row strides are multiples of 8 halfs, arena is 256B
// aligned).  K tail (K%32, only K=2000 here) is peeled into one guarded
// step so the main loop has no branches.
// ---------------------------------------------------------------------------
struct GArgs {
  const _Float16* A;  ll a_rs, a_cs, a_b1, a_b2;
  const _Float16* Bm; ll b_rs, b_cs, b_b1, b_b2;
  float* C;          ll c_b1,  c_b2;  int c_rs;
  _Float16* C16;     ll c16_b1, c16_b2; int c16_rs;
  const float* bias;
  const float* res;  ll r_b1, r_b2; int r_rs;
  int M, N, K, zdiv, Z, relu;
  float alpha;
};

template <bool AKC, bool BKC>
__global__ __launch_bounds__(128) void wmma_gemm_kernel(GArgs g) {
  const int lane = threadIdx.x & 31;
  const int wave = threadIdx.x >> 5;
  const int grp  = lane >> 4;      // half-wave selector
  const int lr   = lane & 15;
  const int m0   = blockIdx.y << 4;
  const int n0   = ((blockIdx.x << 2) + wave) << 4;
  if (n0 >= g.N) return;           // wave-uniform exit (EXEC stays all-ones)

  const int z  = blockIdx.z;
  const int z1 = z / g.zdiv;
  const int z2 = z - z1 * g.zdiv;
  const _Float16* Ap = g.A  + (ll)z1 * g.a_b1 + (ll)z2 * g.a_b2;
  const _Float16* Bp = g.Bm + (ll)z1 * g.b_b1 + (ll)z2 * g.b_b2;

  int mA = m0 + lr; if (mA >= g.M) mA = g.M - 1;   // clamp (loads only)
  int nB = n0 + lr; if (nB >= g.N) nB = g.N - 1;
  const _Float16* pa = Ap + (ll)mA * g.a_rs;   // + k * a_cs
  const _Float16* pb = Bp + (ll)nB * g.b_cs;   // + k * b_rs

  v8f acc = {0.f, 0.f, 0.f, 0.f, 0.f, 0.f, 0.f, 0.f};
  const int kMain = g.K & ~31;
  int k0 = 0;
  // ---- main loop: no bounds checks, vector loads where contiguous ----
  for (; k0 < kMain; k0 += 32) {
    v16h af, bf;
    if (AKC) {
      // A element e -> k = k0 + 8*grp + (e&7) + 16*(e>>3): two 16B chunks
      const _Float16* p = pa + k0 + (grp << 3);
      const v8h lo = *(const v8h*)(p);
      const v8h hi = *(const v8h*)(p + 16);
#pragma unroll
      for (int e = 0; e < 8; ++e) { af[e] = lo[e]; af[e + 8] = hi[e]; }
    } else {
#pragma unroll
      for (int e = 0; e < 16; ++e) {
        const int ka = k0 + ((e & 8) << 1) + (grp << 3) + (e & 7);
        af[e] = pa[(ll)ka * g.a_cs];
      }
    }
    if (BKC) {
      // B element e -> k = k0 + 8*grp + (e&7) + 16*(e>>3): two 16B chunks
      const _Float16* p = pb + k0 + (grp << 3);
      const v8h lo = *(const v8h*)(p);
      const v8h hi = *(const v8h*)(p + 16);
#pragma unroll
      for (int e = 0; e < 8; ++e) { bf[e] = lo[e]; bf[e + 8] = hi[e]; }
    } else {
#pragma unroll
      for (int e = 0; e < 16; ++e)
        bf[e] = pb[(ll)(k0 + ((e & 8) << 1) + (grp << 3) + (e & 7)) * g.b_rs];
    }
    acc = __builtin_amdgcn_wmma_f32_16x16x32_f16(false, af, false, bf,
                                                 (short)0, acc, false, false);
  }
  // ---- peeled ragged tail (K % 32 != 0): guarded scalar loads ----
  if (k0 < g.K) {
    v16h af, bf;
#pragma unroll
    for (int e = 0; e < 16; ++e) {
      const int ka = k0 + ((e & 8) << 1) + (grp << 3) + (e & 7);
      const int kb = k0 + ((e & 8) << 1) + (grp << 3) + (e & 7);
      af[e] = (ka < g.K) ? pa[(ll)ka * g.a_cs] : (_Float16)0.f;
      bf[e] = (kb < g.K) ? pb[(ll)kb * g.b_rs] : (_Float16)0.f;
    }
    acc = __builtin_amdgcn_wmma_f32_16x16x32_f16(false, af, false, bf,
                                                 (short)0, acc, false, false);
  }

  const int n = n0 + lr;
  if (n >= g.N) return;
#pragma unroll
  for (int r = 0; r < 8; ++r) {
    const int m = m0 + r + (grp << 3);        // C layout: VGPR r -> M = r + 8*grp
    if (m >= g.M) continue;
    float v = acc[r] * g.alpha;
    if (g.bias) v += g.bias[m];
    if (g.res)  v += g.res[(ll)z1 * g.r_b1 + (ll)z2 * g.r_b2 + (ll)m * g.r_rs + n];
    if (g.relu) v = fmaxf(v, 0.f);
    if (g.C)    g.C  [(ll)z1 * g.c_b1   + (ll)z2 * g.c_b2   + (ll)m * g.c_rs   + n] = v;
    if (g.C16)  g.C16[(ll)z1 * g.c16_b1 + (ll)z2 * g.c16_b2 + (ll)m * g.c16_rs + n] = (_Float16)v;
  }
}

// ---------------------------------------------------------------------------
// Elementwise / reduction helper kernels (all 256-thread blocks)
// ---------------------------------------------------------------------------
__device__ __forceinline__ float bsum256(float v, float* sh) {
  const int t = threadIdx.x;
  sh[t] = v; __syncthreads();
  for (int s = 128; s > 0; s >>= 1) { if (t < s) sh[t] += sh[t + s]; __syncthreads(); }
  const float r = sh[0]; __syncthreads();
  return r;
}

__global__ void f32_to_f16(const float* __restrict__ s, _Float16* __restrict__ d, ll n) {
  ll i = (ll)blockIdx.x * blockDim.x + threadIdx.x;
  const ll st = (ll)gridDim.x * blockDim.x;
  for (; i < n; i += st) d[i] = (_Float16)s[i];
}

// InstanceNorm over N per (b,c); grid=(C,B), eps=1e-3
__global__ void inorm_kernel(const float* __restrict__ x, float* __restrict__ o,
                             int n, float eps) {
  __shared__ float sh[256];
  const ll base = ((ll)blockIdx.y * gridDim.x + blockIdx.x) * n;
  float s = 0.f, s2 = 0.f;
  for (int i = threadIdx.x; i < n; i += 256) { const float v = x[base + i]; s += v; s2 += v * v; }
  s  = bsum256(s, sh);
  s2 = bsum256(s2, sh);
  const float mean = s / n;
  const float iv = rsqrtf(s2 / n - mean * mean + eps);
  for (int i = threadIdx.x; i < n; i += 256) o[base + i] = (x[base + i] - mean) * iv;
}

// BatchNorm stats over (B, L) per channel; grid = channels
__global__ void bn_stats(const float* __restrict__ x, float* __restrict__ mu,
                         float* __restrict__ iv, int cc, int L, int bn, float eps) {
  __shared__ float sh[256];
  const int ch = blockIdx.x;
  const ll tot = (ll)bn * L;
  float s = 0.f, s2 = 0.f;
  for (ll i = threadIdx.x; i < tot; i += 256) {
    const ll b = i / L, l = i - b * L;
    const float v = x[(b * cc + ch) * (ll)L + l];
    s += v; s2 += v * v;
  }
  s  = bsum256(s, sh);
  s2 = bsum256(s2, sh);
  if (threadIdx.x == 0) {
    const float m = s / (float)tot;
    mu[ch] = m;
    iv[ch] = rsqrtf(s2 / (float)tot - m * m + eps);
  }
}

__global__ void bn_apply_relu(const float* __restrict__ x, const float* __restrict__ mu,
                              const float* __restrict__ iv, const float* __restrict__ gam,
                              const float* __restrict__ bet, _Float16* __restrict__ y,
                              int cc, int L, ll n) {
  ll i = (ll)blockIdx.x * blockDim.x + threadIdx.x;
  const ll st = (ll)gridDim.x * blockDim.x;
  for (; i < n; i += st) {
    const int ch = (int)((i / L) % cc);
    const float v = (x[i] - mu[ch]) * iv[ch] * gam[ch] + bet[ch];
    y[i] = (_Float16)fmaxf(v, 0.f);
  }
}

// Row softmax (f32 in, f16 out); one block per row
__global__ void softmax_rows(const float* __restrict__ X, _Float16* __restrict__ Y, int len) {
  __shared__ float sh[256];
  const ll row = blockIdx.x;
  const float* x = X + row * (ll)len;
  _Float16* y = Y + row * (ll)len;
  const int t = threadIdx.x;
  float m = -3.4e38f;
  for (int i = t; i < len; i += 256) m = fmaxf(m, x[i]);
  sh[t] = m; __syncthreads();
  for (int s = 128; s > 0; s >>= 1) { if (t < s) sh[t] = fmaxf(sh[t], sh[t + s]); __syncthreads(); }
  const float mx = sh[0]; __syncthreads();
  float s = 0.f;
  for (int i = t; i < len; i += 256) s += __expf(x[i] - mx);
  s = bsum256(s, sh);
  const float iv = 1.f / s;
  for (int i = t; i < len; i += 256) y[i] = (_Float16)(__expf(x[i] - mx) * iv);
}

// Copy m1 (f16) into first C rows of the 2C-channel concat buffer
__global__ void copy_into_cat(const _Float16* __restrict__ s, _Float16* __restrict__ d,
                              int cc, int L, ll n) {
  ll i = (ll)blockIdx.x * blockDim.x + threadIdx.x;
  const ll st = (ll)gridDim.x * blockDim.x;
  const ll per = (ll)cc * L;
  for (; i < n; i += st) {
    const ll b = i / per, r = i - b * per;
    d[b * 2 * per + r] = s[i];
  }
}

// feats_weight: relu(BN(feats)) -> dot with fw_w -> sigmoid -> clamp
__global__ void fw_kernel(const float* __restrict__ x, const float* __restrict__ mu,
                          const float* __restrict__ iv, const float* __restrict__ gam,
                          const float* __restrict__ bet, const float* __restrict__ wv,
                          const float* __restrict__ wb, float* __restrict__ wf) {
  const int i = blockIdx.x * 256 + threadIdx.x;
  if (i >= B_ * M_) return;
  const int b = i / M_, m = i - b * M_;
  float acc = wb[0];
  for (int c = 0; c < C_; ++c) {
    float v = x[((ll)b * C_ + c) * M_ + m];
    v = fmaxf((v - mu[c]) * iv[c] * gam[c] + bet[c], 0.f);
    acc += wv[c] * v;
  }
  const float s = 1.f / (1.f + __expf(-acc));
  wf[i] = fminf(0.95f, fmaxf(0.05f, s));
}

// ker[m,n] = exp(-beta*max(G[m,m]+G[n,n]-2G[m,n],0)) from Gram matrix G=pp^T pp
__global__ void ker_build(const float* __restrict__ G, float* __restrict__ ker,
                          _Float16* __restrict__ ker16, const float* __restrict__ beta,
                          int Mm, ll n) {
  ll i = (ll)blockIdx.x * blockDim.x + threadIdx.x;
  const ll st = (ll)gridDim.x * blockDim.x;
  const ll per = (ll)Mm * Mm;
  for (; i < n; i += st) {
    const ll b = i / per; const int r = (int)(i - b * per);
    const int m = r / Mm, nn = r - m * Mm;
    const float* Gb = G + b * per;
    float d2 = Gb[(ll)m * Mm + m] + Gb[(ll)nn * Mm + nn] - 2.f * Gb[(ll)m * Mm + nn];
    const float kv = __expf(-fmaxf(d2, 0.f) * beta[0]);
    ker[i] = kv; ker16[i] = (_Float16)kv;
  }
}

// A_sys[m,n] = wf[m]*wf[n]*ker[m,n] + lamda*delta(m,n)
__global__ void sys_build(const float* __restrict__ ker, const float* __restrict__ wf,
                          const float* __restrict__ lamda, float* __restrict__ A,
                          int Mm, ll n) {
  ll i = (ll)blockIdx.x * blockDim.x + threadIdx.x;
  const ll st = (ll)gridDim.x * blockDim.x;
  const ll per = (ll)Mm * Mm;
  for (; i < n; i += st) {
    const ll b = i / per; const int r = (int)(i - b * per);
    const int m = r / Mm, nn = r - m * Mm;
    float v = wf[b * Mm + m] * wf[b * Mm + nn] * ker[i];
    if (m == nn) v += lamda[0];
    A[i] = v;
  }
}

// Rhs[b,m,c] = wf[b,m] * val[b,c,m]
__global__ void rhs_build(const float* __restrict__ val, const float* __restrict__ wf,
                          float* __restrict__ R, int Mm, int cc, ll n) {
  ll i = (ll)blockIdx.x * blockDim.x + threadIdx.x;
  const ll st = (ll)gridDim.x * blockDim.x;
  const ll per = (ll)Mm * cc;
  for (; i < n; i += st) {
    const ll b = i / per; const int r = (int)(i - b * per);
    const int m = r / cc, c = r - m * cc;
    R[i] = wf[b * Mm + m] * val[((ll)b * cc + c) * Mm + m];
  }
}

// In-place Gauss-Jordan solve of (A | R) per batch.  A is SPD (wkw + lamda*I).
__global__ void solve_kernel(float* __restrict__ A, float* __restrict__ R, int Mm, int cc) {
  const int b = blockIdx.x;
  float* Ab = A + (ll)b * Mm * Mm;
  float* Rb = R + (ll)b * Mm * cc;
  const int t = threadIdx.x;
  for (int k = 0; k < Mm; ++k) {
    const float ip = 1.f / Ab[(ll)k * Mm + k];
    __syncthreads();
    for (int j = k + t; j < Mm; j += blockDim.x) Ab[(ll)k * Mm + j] *= ip;
    for (int j = t;     j < cc; j += blockDim.x) Rb[(ll)k * cc + j] *= ip;
    __syncthreads();
    for (int i = t; i < Mm; i += blockDim.x) {
      if (i == k) continue;
      const float f = Ab[(ll)i * Mm + k];
      if (f == 0.f) continue;
      for (int j = k; j < Mm; ++j) Ab[(ll)i * Mm + j] -= f * Ab[(ll)k * Mm + j];
      for (int j = 0; j < cc; ++j) Rb[(ll)i * cc + j] -= f * Rb[(ll)k * cc + j];
    }
    __syncthreads();
  }
}

// ---------------------------------------------------------------------------
// Host-side plumbing
// ---------------------------------------------------------------------------
static inline GArgs gz() { GArgs g; memset(&g, 0, sizeof(g)); g.zdiv = 1; g.Z = 1; g.alpha = 1.f; return g; }
static inline void launch_gemm(hipStream_t st, const GArgs& g) {
  dim3 grid((unsigned)((g.N + 63) / 64), (unsigned)((g.M + 15) / 16), (unsigned)g.Z);
  const dim3 blk(128);
  const bool akc = (g.a_cs == 1), bkc = (g.b_rs == 1);
  if (akc && bkc)      wmma_gemm_kernel<true,  true ><<<grid, blk, 0, st>>>(g);
  else if (akc)        wmma_gemm_kernel<true,  false><<<grid, blk, 0, st>>>(g);
  else if (bkc)        wmma_gemm_kernel<false, true ><<<grid, blk, 0, st>>>(g);
  else                 wmma_gemm_kernel<false, false><<<grid, blk, 0, st>>>(g);
}
static inline int eb(ll n) { ll b = (n + 255) / 256; if (b > 8192) b = 8192; if (b < 1) b = 1; return (int)b; }

struct AttnP {
  const float *bnb, *bng, *c1b, *c2b, *kb, *mb, *qb, *vb;
  const _Float16 *qw, *kw, *vw, *mw, *c1w, *c2w;
};
struct Scr {
  _Float16 *Q16, *K16, *V16, *A2, *cat16, *h2, *S16;
  float *S32, *H1, *mu, *inv;
};

// Full AttentionPropagation: m1[B,C,L1] <- cross-attn from m2[B,C,L2], gated MLP.
static void run_attn(hipStream_t st, const AttnP& w,
                     const float* m1_32, const _Float16* m1_16, const _Float16* m2_16,
                     int L1, int L2, float* out32, _Float16* out16, const Scr& s) {
  GArgs g;
  // Q = qw @ m1 + qb    (f16 only)
  g = gz(); g.A = w.qw; g.a_rs = C_; g.a_cs = 1;
  g.Bm = m1_16; g.b_rs = L1; g.b_cs = 1; g.b_b1 = (ll)C_ * L1;
  g.C16 = s.Q16; g.c16_b1 = (ll)C_ * L1; g.c16_rs = L1; g.bias = w.qb;
  g.M = C_; g.N = L1; g.K = C_; g.Z = B_;
  launch_gemm(st, g);
  // K = kw @ m2 + kb
  g = gz(); g.A = w.kw; g.a_rs = C_; g.a_cs = 1;
  g.Bm = m2_16; g.b_rs = L2; g.b_cs = 1; g.b_b1 = (ll)C_ * L2;
  g.C16 = s.K16; g.c16_b1 = (ll)C_ * L2; g.c16_rs = L2; g.bias = w.kb;
  g.M = C_; g.N = L2; g.K = C_; g.Z = B_;
  launch_gemm(st, g);
  // V = vw @ m2 + vb
  g = gz(); g.A = w.vw; g.a_rs = C_; g.a_cs = 1;
  g.Bm = m2_16; g.b_rs = L2; g.b_cs = 1; g.b_b1 = (ll)C_ * L2;
  g.C16 = s.V16; g.c16_b1 = (ll)C_ * L2; g.c16_rs = L2; g.bias = w.vb;
  g.M = C_; g.N = L2; g.K = C_; g.Z = B_;
  launch_gemm(st, g);
  // S[b,h,n,m] = (1/sqrt(HD)) * sum_d Q[h*HD+d,n] K[h*HD+d,m]
  g = gz();
  g.A = s.Q16; g.a_rs = 1; g.a_cs = L1; g.a_b1 = (ll)C_ * L1; g.a_b2 = (ll)HD_ * L1;
  g.Bm = s.K16; g.b_rs = L2; g.b_cs = 1; g.b_b1 = (ll)C_ * L2; g.b_b2 = (ll)HD_ * L2;
  g.C = s.S32; g.c_b1 = (ll)H_ * L1 * L2; g.c_b2 = (ll)L1 * L2; g.c_rs = L2;
  g.M = L1; g.N = L2; g.K = HD_; g.zdiv = H_; g.Z = B_ * H_;
  g.alpha = 0.17677669529663687f;  // 1/sqrt(32)
  launch_gemm(st, g);
  softmax_rows<<<B_ * H_ * L1, 256, 0, st>>>(s.S32, s.S16, L2);
  // A2[b, h*HD+d, n] = sum_m S[n,m] V[h*HD+d, m]   (f16 only)
  g = gz();
  g.A = s.V16; g.a_rs = L2; g.a_cs = 1; g.a_b1 = (ll)C_ * L2; g.a_b2 = (ll)HD_ * L2;
  g.Bm = s.S16; g.b_rs = 1; g.b_cs = L2; g.b_b1 = (ll)H_ * L1 * L2; g.b_b2 = (ll)L1 * L2;
  g.C16 = s.A2; g.c16_b1 = (ll)C_ * L1; g.c16_b2 = (ll)HD_ * L1; g.c16_rs = L1;
  g.M = HD_; g.N = L1; g.K = L2; g.zdiv = H_; g.Z = B_ * H_;
  launch_gemm(st, g);
  // cat = [m1 ; mw@A2+mb]  (2C channels, f16)
  const ll ncl = (ll)B_ * C_ * L1;
  copy_into_cat<<<eb(ncl), 256, 0, st>>>(m1_16, s.cat16, C_, L1, ncl);
  g = gz(); g.A = w.mw; g.a_rs = C_; g.a_cs = 1;
  g.Bm = s.A2; g.b_rs = L1; g.b_cs = 1; g.b_b1 = (ll)C_ * L1;
  g.C16 = s.cat16 + (ll)C_ * L1; g.c16_b1 = (ll)C2_ * L1; g.c16_rs = L1; g.bias = w.mb;
  g.M = C_; g.N = L1; g.K = C_; g.Z = B_;
  launch_gemm(st, g);
  // H1 = c1w @ cat + c1b  (f32 for BN stats)
  g = gz(); g.A = w.c1w; g.a_rs = C2_; g.a_cs = 1;
  g.Bm = s.cat16; g.b_rs = L1; g.b_cs = 1; g.b_b1 = (ll)C2_ * L1;
  g.C = s.H1; g.c_b1 = (ll)C2_ * L1; g.c_rs = L1; g.bias = w.c1b;
  g.M = C2_; g.N = L1; g.K = C2_; g.Z = B_;
  launch_gemm(st, g);
  bn_stats<<<C2_, 256, 0, st>>>(s.H1, s.mu, s.inv, C2_, L1, B_, 1e-5f);
  const ll nh = (ll)B_ * C2_ * L1;
  bn_apply_relu<<<eb(nh), 256, 0, st>>>(s.H1, s.mu, s.inv, w.bng, w.bnb, s.h2, C2_, L1, nh);
  // out = m1 + c2w @ relu(BN) + c2b
  g = gz(); g.A = w.c2w; g.a_rs = C2_; g.a_cs = 1;
  g.Bm = s.h2; g.b_rs = L1; g.b_cs = 1; g.b_b1 = (ll)C2_ * L1;
  g.C = out32; g.c_b1 = (ll)C_ * L1; g.c_rs = L1;
  g.C16 = out16; g.c16_b1 = (ll)C_ * L1; g.c16_rs = L1;
  g.bias = w.c2b; g.res = m1_32; g.r_b1 = (ll)C_ * L1; g.r_rs = L1;
  g.M = C_; g.N = L1; g.K = C2_; g.Z = B_;
  launch_gemm(st, g);
}

// ---------------------------------------------------------------------------
// kernel_launch
// Input order: pos, corr_feats, then params flattened jax-tree style
// (dict keys sorted alphabetically, recursively):
//   2 beta | 3 fw_b | 4 fw_bnb | 5 fw_bng | 6 fw_w |
//   7..20 inject{bnb,bng,c1b,c1w,c2b,c2w,kb,kw,mb,mw,qb,qw,vb,vw} |
//   21 lamda | 22 pos_b | 23 pos_w | 24..37 rect{...} | 38..51 rect1{...} |
//   52 samp_b | 53 samp_bnb | 54 samp_bng | 55 samp_w | 56 val_b | 57 val_w
// ---------------------------------------------------------------------------
extern "C" void kernel_launch(void* const* d_in, const int* in_sizes, int n_in,
                              void* d_out, int out_size, void* d_ws, size_t ws_size,
                              hipStream_t stream) {
  (void)in_sizes; (void)n_in; (void)out_size; (void)ws_size;
  auto Pf = [&](int i) -> const float* { return (const float*)d_in[i]; };
  const float* pos = Pf(0);   // [B, C/2, N]
  const float* F   = Pf(1);   // [B, C,   N]

  // ---- scratch arena (bump allocator; regions reused across phases) ----
  char* basep = (char*)d_ws;
  size_t off = 0;
  auto af = [&](ll n) -> float* {
    float* p = (float*)(basep + off); off += (size_t)n * 4; off = (off + 255) & ~(size_t)255; return p; };
  auto ah = [&](ll n) -> _Float16* {
    _Float16* p = (_Float16*)(basep + off); off += (size_t)n * 2; off = (off + 255) & ~(size_t)255; return p; };

  float*    S32   = af((ll)B_ * H_ * M_ * N_);   // scores / sampling logits
  _Float16* S16   = ah((ll)B_ * H_ * M_ * N_);   // softmaxed scores / Wm16
  float*    H1    = af((ll)B_ * C2_ * N_);       // attn hidden f32 (also inorm h32)
  _Float16* h2    = ah((ll)B_ * C2_ * N_);       // attn hidden f16 (also sampling h16)
  _Float16* Q16   = ah((ll)B_ * C_ * N_);
  _Float16* K16   = ah((ll)B_ * C_ * N_);
  _Float16* V16   = ah((ll)B_ * C_ * N_);
  _Float16* A2    = ah((ll)B_ * C_ * N_);
  _Float16* cat16 = ah((ll)B_ * C2_ * N_);
  _Float16* F16   = ah((ll)B_ * C_ * N_);
  _Float16* P16   = ah((ll)B_ * 64 * N_);
  float*    feats32   = af((ll)B_ * C_ * M_);
  _Float16* feats16   = ah((ll)B_ * C_ * M_);
  float*    feats2_32 = af((ll)B_ * C_ * M_);
  _Float16* feats2_16 = ah((ll)B_ * C_ * M_);
  _Float16* posm16 = ah((ll)B_ * 64 * M_);
  _Float16* pp16   = ah((ll)B_ * 64 * M_);
  float*    G32    = af((ll)B_ * M_ * M_);
  float*    ker32  = af((ll)B_ * M_ * M_);
  _Float16* ker16  = ah((ll)B_ * M_ * M_);
  float*    Asys   = af((ll)B_ * M_ * M_);
  float*    Rhs    = af((ll)B_ * M_ * C_);
  _Float16* Csol16 = ah((ll)B_ * M_ * C_);
  float*    val32  = af((ll)B_ * C_ * M_);
  float*    wfv    = af((ll)B_ * M_);
  float*    pre32  = af((ll)B_ * C_ * M_);
  _Float16* pre16  = ah((ll)B_ * C_ * M_);
  float*    pre2_32 = af((ll)B_ * C_ * M_);
  _Float16* pre2_16 = ah((ll)B_ * C_ * M_);
  float*    mu  = af(C2_);
  float*    inv = af(C2_);

  auto cvt = [&](const float* s, _Float16* d, ll n) {
    f32_to_f16<<<eb(n), 256, 0, stream>>>(s, d, n);
  };

  // ---- f16 weight copies ----
  _Float16* samp_w16 = ah((ll)M_ * C_);  cvt(Pf(55), samp_w16, (ll)M_ * C_);
  _Float16* pos_w16  = ah(64 * 64);      cvt(Pf(23), pos_w16, 64 * 64);
  _Float16* val_w16  = ah((ll)C_ * C_);  cvt(Pf(57), val_w16, (ll)C_ * C_);
  auto load_attn = [&](int b0) -> AttnP {
    AttnP w;
    w.bnb = Pf(b0 + 0); w.bng = Pf(b0 + 1); w.c1b = Pf(b0 + 2);
    const float* c1w = Pf(b0 + 3); w.c2b = Pf(b0 + 4); const float* c2w = Pf(b0 + 5);
    w.kb = Pf(b0 + 6); const float* kw = Pf(b0 + 7);
    w.mb = Pf(b0 + 8); const float* mw = Pf(b0 + 9);
    w.qb = Pf(b0 + 10); const float* qw = Pf(b0 + 11);
    w.vb = Pf(b0 + 12); const float* vw = Pf(b0 + 13);
    _Float16* t;
    t = ah((ll)C_ * C_);   cvt(qw, t, (ll)C_ * C_);   w.qw = t;
    t = ah((ll)C_ * C_);   cvt(kw, t, (ll)C_ * C_);   w.kw = t;
    t = ah((ll)C_ * C_);   cvt(vw, t, (ll)C_ * C_);   w.vw = t;
    t = ah((ll)C_ * C_);   cvt(mw, t, (ll)C_ * C_);   w.mw = t;
    t = ah((ll)C2_ * C2_); cvt(c1w, t, (ll)C2_ * C2_); w.c1w = t;
    t = ah((ll)C_ * C2_);  cvt(c2w, t, (ll)C_ * C2_);  w.c2w = t;
    return w;
  };
  AttnP Winj = load_attn(7);
  AttnP Wr   = load_attn(24);
  AttnP Wr1  = load_attn(38);

  // ---- f16 activations of inputs ----
  cvt(F,   F16, (ll)B_ * C_ * N_);
  cvt(pos, P16, (ll)B_ * 64 * N_);

  Scr scr; scr.Q16 = Q16; scr.K16 = K16; scr.V16 = V16; scr.A2 = A2;
  scr.cat16 = cat16; scr.h2 = h2; scr.S16 = S16; scr.S32 = S32;
  scr.H1 = H1; scr.mu = mu; scr.inv = inv;

  GArgs g;

  // ================= AdaptiveSampling =================
  float* h32 = H1;     // alias (sampling happens before any attention uses H1)
  _Float16* h16 = h2;  // alias
  inorm_kernel<<<dim3(C_, B_), 256, 0, stream>>>(F, h32, N_, 1e-3f);
  bn_stats<<<C_, 256, 0, stream>>>(h32, mu, inv, C_, N_, B_, 1e-5f);
  const ll nhs = (ll)B_ * C_ * N_;
  bn_apply_relu<<<eb(nhs), 256, 0, stream>>>(h32, mu, inv, Pf(54), Pf(53), h16, C_, N_, nhs);
  // logits[b,m,n] = samp_w @ h + samp_b
  g = gz(); g.A = samp_w16; g.a_rs = C_; g.a_cs = 1;
  g.Bm = h16; g.b_rs = N_; g.b_cs = 1; g.b_b1 = (ll)C_ * N_;
  g.C = S32; g.c_b1 = (ll)M_ * N_; g.c_rs = N_; g.bias = Pf(52);
  g.M = M_; g.N = N_; g.K = C_; g.Z = B_;
  launch_gemm(stream, g);
  _Float16* Wm16 = S16;
  softmax_rows<<<B_ * M_, 256, 0, stream>>>(S32, Wm16, N_);
  // feats[b,c,m] = sum_n F[c,n] Wm[m,n]
  g = gz(); g.A = F16; g.a_rs = N_; g.a_cs = 1; g.a_b1 = (ll)C_ * N_;
  g.Bm = Wm16; g.b_rs = 1; g.b_cs = N_; g.b_b1 = (ll)M_ * N_;
  g.C = feats32; g.c_b1 = (ll)C_ * M_; g.c_rs = M_;
  g.C16 = feats16; g.c16_b1 = (ll)C_ * M_; g.c16_rs = M_;
  g.M = C_; g.N = M_; g.K = N_; g.Z = B_;
  launch_gemm(stream, g);
  // pos_m[b,d,m] = sum_n P[d,n] Wm[m,n]  (f16 only)
  g = gz(); g.A = P16; g.a_rs = N_; g.a_cs = 1; g.a_b1 = (ll)64 * N_;
  g.Bm = Wm16; g.b_rs = 1; g.b_cs = N_; g.b_b1 = (ll)M_ * N_;
  g.C16 = posm16; g.c16_b1 = (ll)64 * M_; g.c16_rs = M_;
  g.M = 64; g.N = M_; g.K = N_; g.Z = B_;
  launch_gemm(stream, g);

  // ================= inject attention =================
  run_attn(stream, Winj, feats32, feats16, F16, M_, N_, feats2_32, feats2_16, scr);

  // ================= feats_weight =================
  bn_stats<<<C_, 256, 0, stream>>>(feats2_32, mu, inv, C_, M_, B_, 1e-5f);
  fw_kernel<<<(B_ * M_ + 255) / 256, 256, 0, stream>>>(feats2_32, mu, inv,
      Pf(5), Pf(4), Pf(6), Pf(3), wfv);

  // ================= LearnableKernel + ridge solve =================
  // val = val_w @ feats + val_b
  g = gz(); g.A = val_w16; g.a_rs = C_; g.a_cs = 1;
  g.Bm = feats2_16; g.b_rs = M_; g.b_cs = 1; g.b_b1 = (ll)C_ * M_;
  g.C = val32; g.c_b1 = (ll)C_ * M_; g.c_rs = M_; g.bias = Pf(56);
  g.M = C_; g.N = M_; g.K = C_; g.Z = B_;
  launch_gemm(stream, g);
  // pp = pos_w @ pos_m + pos_b   (f16 only)
  g = gz(); g.A = pos_w16; g.a_rs = 64; g.a_cs = 1;
  g.Bm = posm16; g.b_rs = M_; g.b_cs = 1; g.b_b1 = (ll)64 * M_;
  g.C16 = pp16; g.c16_b1 = (ll)64 * M_; g.c16_rs = M_; g.bias = Pf(22);
  g.M = 64; g.N = M_; g.K = 64; g.Z = B_;
  launch_gemm(stream, g);
  // Gram G[m,n] = sum_d pp[d,m] pp[d,n]
  g = gz(); g.A = pp16; g.a_rs = 1; g.a_cs = M_; g.a_b1 = (ll)64 * M_;
  g.Bm = pp16; g.b_rs = M_; g.b_cs = 1; g.b_b1 = (ll)64 * M_;
  g.C = G32; g.c_b1 = (ll)M_ * M_; g.c_rs = M_;
  g.M = M_; g.N = M_; g.K = 64; g.Z = B_;
  launch_gemm(stream, g);
  const ll nmm = (ll)B_ * M_ * M_;
  const ll nmc = (ll)B_ * M_ * C_;
  ker_build<<<eb(nmm), 256, 0, stream>>>(G32, ker32, ker16, Pf(2), M_, nmm);
  sys_build<<<eb(nmm), 256, 0, stream>>>(ker32, wfv, Pf(21), Asys, M_, nmm);
  rhs_build<<<eb(nmc), 256, 0, stream>>>(val32, wfv, Rhs, M_, C_, nmc);
  solve_kernel<<<B_, 256, 0, stream>>>(Asys, Rhs, M_, C_);
  cvt(Rhs, Csol16, nmc);   // Csol [B, M, C]
  // pre[b,c,m] = sum_n ker[m,n] Csol[n,c]
  g = gz(); g.A = Csol16; g.a_rs = 1; g.a_cs = C_; g.a_b1 = (ll)M_ * C_;
  g.Bm = ker16; g.b_rs = 1; g.b_cs = M_; g.b_b1 = (ll)M_ * M_;
  g.C = pre32; g.c_b1 = (ll)C_ * M_; g.c_rs = M_;
  g.C16 = pre16; g.c16_b1 = (ll)C_ * M_; g.c16_rs = M_;
  g.M = C_; g.N = M_; g.K = M_; g.Z = B_;
  launch_gemm(stream, g);

  // ================= rectify attentions =================
  run_attn(stream, Wr1, feats2_32, feats2_16, pre16, M_, M_, pre2_32, pre2_16, scr);
  run_attn(stream, Wr, F, F16, pre2_16, N_, M_, (float*)d_out, nullptr, scr);
}
